// FDGAND_ABLTEST_Model_38491496906861
// MI455X (gfx1250) — matrix operand order repaired and checked by reading, not compiled, not run on hardware
//
#include <hip/hip_runtime.h>

typedef __attribute__((ext_vector_type(16))) __bf16 v16bf;
typedef __attribute__((ext_vector_type(8)))  float  v8f;

#define ROWS    128      // rows per workgroup (8 waves x 16 rows)
#define THREADS 256
#define AS      520      // bf16 activation LDS row stride (16B-aligned rows)
#define WTS     520      // bf16 W-tile LDS row stride
#define H2S     129      // f32 h2 LDS row stride (bank-conflict-free)

__device__ __forceinline__ unsigned short f2bf(float f) {
    union { float f; unsigned u; } x; x.f = f;
    unsigned r = x.u + 0x7fffu + ((x.u >> 16) & 1u);   // round-to-nearest-even
    return (unsigned short)(r >> 16);
}
__device__ __forceinline__ float bf2f(unsigned short b) {
    union { unsigned u; float f; } x; x.u = ((unsigned)b) << 16;
    return x.f;
}
__device__ __forceinline__ float celu1(float x) {
    return x > 0.f ? x : (expf(x) - 1.f);
}

union Frag { v16bf v; uint4 q[2]; };

__device__ __forceinline__ void loadB(Frag& b, const unsigned short* wrow,
                                      int ks, int half) {
    // B 32x16 bf16: lane holds col n, lanes0-15 K0..7/K16..23, lanes16-31 K8..15/K24..31
    const unsigned short* pb = wrow + ks * 32 + half * 8;
    b.q[0] = *(const uint4*)(pb);
    b.q[1] = *(const uint4*)(pb + 16);
}

// h[ROWS][N] = act[ROWS][K](bf16,LDS) @ W[N][K](f32,global)^T + bias.
// A register-resident for the whole layer; two N-tiles (two C chains) per pass;
// 32-row W tile staged in LDS, double-buffered through registers.
template<int K, int N, bool BF16OUT>
__device__ void gemm_tile(const unsigned short* act, const float* W,
                          unsigned short* wt,
                          unsigned short* destB, float* destF, int ds,
                          const float* bias, int tid, int wid, int lane)
{
    constexpr int KS  = K / 32;                     // k-steps per tile
    constexpr int NT2 = N / 32;                     // n-tile pairs
    constexpr int F4  = (32 * (K / 4)) / THREADS;   // float4/thread per pair (16)
    const int row0 = wid * 16;
    const int nlo  = lane & 15;     // A row / B col / C col within tile
    const int half = lane >> 4;
    const int sr   = tid >> 3;              // staging: 8 threads per W row (32 rows)
    const int sc0  = (tid & 7) * F4;        // starting float4 within row

    __syncthreads();                // act (xs or LN'd h1) ready

    // ---- hoist this wave's A operand (16 rows x K) into registers ----
    Frag a[KS];
    #pragma unroll
    for (int ks = 0; ks < KS; ++ks) {
        // A 16x32 bf16: lanes0-15 K0..7/K16..23, lanes16-31 K8..15/K24..31
        const unsigned short* pa = act + (row0 + nlo) * AS + ks * 32;
        a[ks].q[0] = *(const uint4*)(pa + half * 8);
        a[ks].q[1] = *(const uint4*)(pa + 16 + half * 8);
    }

    // ---- prefetch W tile-pair 0 into registers ----
    float4 wreg[F4];
    #pragma unroll
    for (int j = 0; j < F4; ++j)
        wreg[j] = *(const float4*)(W + (long)sr * K + (sc0 + j) * 4);

    for (int nt2 = 0; nt2 < NT2; ++nt2) {
        const int n0 = nt2 * 32;
        __syncthreads();            // previous pair's B reads done; wt free
        #pragma unroll
        for (int j = 0; j < F4; ++j) {
            uint2 p;
            p.x = (unsigned)f2bf(wreg[j].x) | ((unsigned)f2bf(wreg[j].y) << 16);
            p.y = (unsigned)f2bf(wreg[j].z) | ((unsigned)f2bf(wreg[j].w) << 16);
            *(uint2*)(wt + sr * WTS + (sc0 + j) * 4) = p;
        }
        __syncthreads();            // wt ready

        // overlap: fetch next tile-pair while this one computes
        if (nt2 + 1 < NT2) {
            const float* Wn = W + (long)(nt2 + 1) * 32 * K;
            #pragma unroll
            for (int j = 0; j < F4; ++j)
                wreg[j] = *(const float4*)(Wn + (long)sr * K + (sc0 + j) * 4);
        }
        if (nt2 + 2 < NT2)          // L2 prefetch two pairs ahead
            __builtin_prefetch(W + (long)(nt2 + 2) * 32 * K + tid * 64, 0, 0);

        // ---- two independent accumulation chains, B pipelined by 1 ----
        const unsigned short* wr0 = wt + nlo * WTS;
        const unsigned short* wr1 = wt + (nlo + 16) * WTS;
        v8f c0 = {}, c1 = {};
        Frag b0[2], b1[2];
        loadB(b0[0], wr0, 0, half);
        loadB(b1[0], wr1, 0, half);
        #pragma unroll
        for (int ks = 0; ks < KS; ++ks) {
            if (ks + 1 < KS) {
                loadB(b0[(ks + 1) & 1], wr0, ks + 1, half);
                loadB(b1[(ks + 1) & 1], wr1, ks + 1, half);
            }
            c0 = __builtin_amdgcn_wmma_f32_16x16x32_bf16(false, a[ks].v, false,
                                                         b0[ks & 1].v, (short)0, c0,
                                                         false, false);
            c1 = __builtin_amdgcn_wmma_f32_16x16x32_bf16(false, a[ks].v, false,
                                                         b1[ks & 1].v, (short)0, c1,
                                                         false, false);
        }

        const float bn0 = bias[n0 + nlo];
        const float bn1 = bias[n0 + 16 + nlo];
        #pragma unroll
        for (int v = 0; v < 8; ++v) {
            const int rr = row0 + half * 8 + v;     // C layout: VGPR v -> M = v(+8)
            const float v0 = c0[v] + bn0;
            const float v1 = c1[v] + bn1;
            if constexpr (BF16OUT) {
                destB[rr * ds + n0 + nlo]      = f2bf(v0);
                destB[rr * ds + n0 + 16 + nlo] = f2bf(v1);
            } else {
                destF[rr * ds + n0 + nlo]      = v0;
                destF[rr * ds + n0 + 16 + nlo] = v1;
            }
        }
    }
}

__global__ void __launch_bounds__(THREADS, 2)
fused_mlp_sn(const float* __restrict__ x,
             const float* __restrict__ W1, const float* __restrict__ b1,
             const float* __restrict__ g1, const float* __restrict__ be1,
             const float* __restrict__ W2, const float* __restrict__ b2,
             const float* __restrict__ g2, const float* __restrict__ be2,
             const float* __restrict__ W3, const float* __restrict__ b3,
             const float* __restrict__ g3, const float* __restrict__ be3,
             const float* __restrict__ W4, const float* __restrict__ b4,
             const float* __restrict__ g4, const float* __restrict__ be4,
             const float* __restrict__ Wh, const float* __restrict__ bh,
             const float* __restrict__ u,
             float* __restrict__ out)
{
    extern __shared__ char smem[];
    unsigned short* xs  = (unsigned short*)smem;              // 128x520 bf16 (133120 B)
    unsigned short* h1s = (unsigned short*)(smem + 133120);   // 128x520 bf16 (133120 B)
    unsigned short* wt  = (unsigned short*)(smem + 266240);   // 32x520  bf16 (33280 B)
    float*          h2s = (float*)smem;                       // reuse xs: 128x129 f32

    const int  tid   = threadIdx.x;
    const int  wid   = tid >> 5;
    const int  lane  = tid & 31;
    const long row0g = (long)blockIdx.x * ROWS;

    // ---- load x tile, f32 -> bf16 LDS ----
    for (int i = tid; i < ROWS * 128; i += THREADS) {     // 128 float4 per row
        const int r  = i >> 7;
        const int c4 = i & 127;
        float4 v = *(const float4*)(x + (row0g + r) * 512 + c4 * 4);
        uint2 p;
        p.x = (unsigned)f2bf(v.x) | ((unsigned)f2bf(v.y) << 16);
        p.y = (unsigned)f2bf(v.z) | ((unsigned)f2bf(v.w) << 16);
        *(uint2*)(xs + r * AS + c4 * 4) = p;
    }
    // (entry barrier of gemm_tile covers xs visibility)

    // ---- layer 1: h1 = x @ W1^T + b1 ----
    gemm_tile<512, 512, true>(xs, W1, wt, h1s, nullptr, AS, b1, tid, wid, lane);
    __syncthreads();

    // ---- LN + CELU on h1 (bf16 in place, f32 math) ----
    if (tid < ROWS) {
        unsigned short* row = h1s + tid * AS;
        float mu = 0.f;
        for (int c4 = 0; c4 < 128; ++c4) {
            uint2 pq = *(const uint2*)(row + c4 * 4);
            mu += bf2f((unsigned short)pq.x) + bf2f((unsigned short)(pq.x >> 16))
                + bf2f((unsigned short)pq.y) + bf2f((unsigned short)(pq.y >> 16));
        }
        mu *= (1.f / 512.f);
        float var = 0.f;
        for (int c4 = 0; c4 < 128; ++c4) {
            uint2 pq = *(const uint2*)(row + c4 * 4);
            float d0 = bf2f((unsigned short)pq.x)         - mu;
            float d1 = bf2f((unsigned short)(pq.x >> 16)) - mu;
            float d2 = bf2f((unsigned short)pq.y)         - mu;
            float d3 = bf2f((unsigned short)(pq.y >> 16)) - mu;
            var += d0 * d0 + d1 * d1 + d2 * d2 + d3 * d3;
        }
        const float rs = rsqrtf(var * (1.f / 512.f) + 1e-5f);
        for (int c = 0; c < 512; c += 2) {
            unsigned pq = *(const unsigned*)(row + c);
            float v0 = (bf2f((unsigned short)pq)         - mu) * rs * g1[c]     + be1[c];
            float v1 = (bf2f((unsigned short)(pq >> 16)) - mu) * rs * g1[c + 1] + be1[c + 1];
            *(unsigned*)(row + c) =
                (unsigned)f2bf(celu1(v0)) | ((unsigned)f2bf(celu1(v1)) << 16);
        }
    }

    // ---- layer 2: h2 = h1 @ W2^T + b2 (f32 into reused xs region) ----
    gemm_tile<512, 128, false>(h1s, W2, wt, nullptr, h2s, H2S, b2, tid, wid, lane);
    __syncthreads();

    // ---- LN2 + CELU, then tiny layers 3/4 + spectral-norm head, per row ----
    if (tid < ROWS) {
        float* row = h2s + tid * H2S;
        float mu = 0.f;
        for (int c = 0; c < 128; ++c) mu += row[c];
        mu *= (1.f / 128.f);
        float var = 0.f;
        for (int c = 0; c < 128; ++c) { float d = row[c] - mu; var += d * d; }
        const float rs = rsqrtf(var * (1.f / 128.f) + 1e-5f);
        for (int c = 0; c < 128; ++c)
            row[c] = celu1((row[c] - mu) * rs * g2[c] + be2[c]);

        // layer 3: 128 -> 32
        float h3[32];
        #pragma unroll
        for (int j = 0; j < 32; ++j) {
            float acc = b3[j];
            for (int k = 0; k < 128; ++k) acc += row[k] * W3[j * 128 + k];
            h3[j] = acc;
        }
        float mu3 = 0.f;
        #pragma unroll
        for (int j = 0; j < 32; ++j) mu3 += h3[j];
        mu3 *= (1.f / 32.f);
        float var3 = 0.f;
        #pragma unroll
        for (int j = 0; j < 32; ++j) { float d = h3[j] - mu3; var3 += d * d; }
        const float rs3 = rsqrtf(var3 * (1.f / 32.f) + 1e-5f);
        #pragma unroll
        for (int j = 0; j < 32; ++j)
            h3[j] = celu1((h3[j] - mu3) * rs3 * g3[j] + be3[j]);

        // layer 4: 32 -> 8
        float h4[8];
        #pragma unroll
        for (int j = 0; j < 8; ++j) {
            float acc = b4[j];
            #pragma unroll
            for (int k = 0; k < 32; ++k) acc += h3[k] * W4[j * 32 + k];
            h4[j] = acc;
        }
        float mu4 = 0.f;
        #pragma unroll
        for (int j = 0; j < 8; ++j) mu4 += h4[j];
        mu4 *= 0.125f;
        float var4 = 0.f;
        #pragma unroll
        for (int j = 0; j < 8; ++j) { float d = h4[j] - mu4; var4 += d * d; }
        const float rs4 = rsqrtf(var4 * 0.125f + 1e-5f);
        #pragma unroll
        for (int j = 0; j < 8; ++j)
            h4[j] = celu1((h4[j] - mu4) * rs4 * g4[j] + be4[j]);

        // spectral-norm head: Wsn = Wh / sigma (one power iteration)
        const float u0 = u[0];
        float vv[8], nv = 0.f;
        #pragma unroll
        for (int j = 0; j < 8; ++j) { vv[j] = Wh[j] * u0; nv += vv[j] * vv[j]; }
        nv = sqrtf(nv) + 1e-12f;
        float s = 0.f;
        #pragma unroll
        for (int j = 0; j < 8; ++j) s += Wh[j] * (vv[j] / nv);
        const float u2    = s / (sqrtf(s * s) + 1e-12f);
        const float sigma = u2 * s;

        float acc = 0.f;
        #pragma unroll
        for (int j = 0; j < 8; ++j) acc += h4[j] * Wh[j];
        out[row0g + tid] = acc / sigma + bh[0];
    }
}

extern "C" void kernel_launch(void* const* d_in, const int* in_sizes, int n_in,
                              void* d_out, int out_size, void* d_ws, size_t ws_size,
                              hipStream_t stream) {
    (void)n_in; (void)d_ws; (void)ws_size; (void)out_size;
    const float* x   = (const float*)d_in[0];
    const float* W1  = (const float*)d_in[1];
    const float* b1  = (const float*)d_in[2];
    const float* g1  = (const float*)d_in[3];
    const float* be1 = (const float*)d_in[4];
    const float* W2  = (const float*)d_in[5];
    const float* b2  = (const float*)d_in[6];
    const float* g2  = (const float*)d_in[7];
    const float* be2 = (const float*)d_in[8];
    const float* W3  = (const float*)d_in[9];
    const float* b3  = (const float*)d_in[10];
    const float* g3  = (const float*)d_in[11];
    const float* be3 = (const float*)d_in[12];
    const float* W4  = (const float*)d_in[13];
    const float* b4  = (const float*)d_in[14];
    const float* g4  = (const float*)d_in[15];
    const float* be4 = (const float*)d_in[16];
    const float* Wh  = (const float*)d_in[17];
    const float* bh  = (const float*)d_in[18];
    const float* u   = (const float*)d_in[19];
    float* out = (float*)d_out;

    const int Brows = in_sizes[0] / 512;          // 131072
    const int grid  = Brows / ROWS;               // 1024
    const size_t smem = 266240 + 32 * WTS * 2;    // 299,520 B < 320 KB

    hipFuncSetAttribute((const void*)fused_mlp_sn,
                        hipFuncAttributeMaxDynamicSharedMemorySize, (int)smem);
    fused_mlp_sn<<<grid, THREADS, smem, stream>>>(
        x, W1, b1, g1, be1, W2, b2, g2, be2, W3, b3, g3, be3,
        W4, b4, g4, be4, Wh, bh, u, out);
}
